// UPTRec_53317724012942
// MI455X (gfx1250) — matrix operand order repaired and checked by reading, not compiled, not run on hardware
//
#include <hip/hip_runtime.h>
#include <hip/hip_bf16.h>
#include <math.h>

// ---------------- problem constants ----------------
#define BB      128
#define TT      200
#define CC      512
#define NH      8
#define HD      64
#define NCLUS   10
#define KM_IT   20
#define BT      (BB * TT)          // 25600
#define BTC     ((size_t)BT * CC)  // 13,107,200 floats

typedef _Float16 v16h __attribute__((ext_vector_type(16)));
typedef _Float16 h8   __attribute__((ext_vector_type(8)));
typedef _Float16 h2   __attribute__((ext_vector_type(2)));
typedef float    v8f  __attribute__((ext_vector_type(8)));
typedef float    f4   __attribute__((ext_vector_type(4)));

// ---------------- f32 -> f16 weight conversion WITH transpose ----------------
// dst[n*512 + k] = (f16) src[k*512 + n]  (so GEMM B-staging is contiguous in k)
__global__ __launch_bounds__(256) void cvt_f16t_kernel(const float* __restrict__ src,
                                                       _Float16* __restrict__ dst) {
    __shared__ float tile[32][33];
    int k0 = blockIdx.x * 32, n0 = blockIdx.y * 32;
    int tx = threadIdx.x & 31, ty = threadIdx.x >> 5;   // 32 x 8
#pragma unroll
    for (int i = 0; i < 4; ++i) {
        int k = ty + i * 8;
        tile[k][tx] = src[(size_t)(k0 + k) * CC + n0 + tx];
    }
    __syncthreads();
#pragma unroll
    for (int i = 0; i < 4; ++i) {
        int n = ty + i * 8;
        dst[(size_t)(n0 + n) * CC + k0 + tx] = (_Float16)tile[tx][n];
    }
}

// ---------------- embedding gather + concat + pos emb ----------------
__global__ __launch_bounds__(256) void embed_kernel(const int* __restrict__ seq,
                                                    const int* __restrict__ uids,
                                                    const float* __restrict__ item_emb,
                                                    const float* __restrict__ user_emb,
                                                    const float* __restrict__ pos_emb,
                                                    float* __restrict__ x) {
    size_t idx = (size_t)blockIdx.x * 256 + threadIdx.x;   // over B*T*C
    int    c   = (int)(idx & (CC - 1));
    size_t bt  = idx >> 9;
    int    t   = (int)(bt % TT);
    int    b   = (int)(bt / TT);
    float  e;
    if (c < 256) e = item_emb[(size_t)seq[bt] * 256 + c];
    else         e = user_emb[(size_t)uids[b] * 256 + (c - 256)];
    x[idx] = e + pos_emb[(size_t)t * CC + c];
}

// ---------------- elementwise mask / residual+mask ----------------
__global__ __launch_bounds__(256) void mask_kernel(float* __restrict__ x,
                                                   const int* __restrict__ seq) {
    size_t idx = (size_t)blockIdx.x * 256 + threadIdx.x;
    size_t bt  = idx >> 9;
    if (seq[bt] == 0) x[idx] = 0.f;
}

__global__ __launch_bounds__(256) void addmask_kernel(const float* __restrict__ a,
                                                      const float* __restrict__ bsrc,
                                                      float* __restrict__ x,
                                                      const int* __restrict__ seq) {
    size_t idx = (size_t)blockIdx.x * 256 + threadIdx.x;
    size_t bt  = idx >> 9;
    float  v   = a[idx] + bsrc[idx];
    x[idx] = (seq[bt] != 0) ? v : 0.f;
}

// ---------------- layernorm: 8 rows per block, one wave32 per row ----------------
__global__ __launch_bounds__(256) void ln_kernel(const float* __restrict__ X,
                                                 float* __restrict__ Y,
                                                 const float* __restrict__ g,
                                                 const float* __restrict__ be) {
    int w = threadIdx.x >> 5, lane = threadIdx.x & 31;
    size_t row = (size_t)blockIdx.x * 8 + w;
    const float* x = X + row * CC;
    float v[16], s = 0.f, ss = 0.f;
#pragma unroll
    for (int i = 0; i < 16; ++i) {
        float t = x[lane + 32 * i];
        v[i] = t; s += t; ss += t * t;
    }
#pragma unroll
    for (int o = 16; o > 0; o >>= 1) {
        s  += __shfl_xor(s,  o, 32);
        ss += __shfl_xor(ss, o, 32);
    }
    float mean = s * (1.f / CC);
    float var  = ss * (1.f / CC) - mean * mean;
    float inv  = rsqrtf(var + 1e-8f);
    float* y = Y + row * CC;
#pragma unroll
    for (int i = 0; i < 16; ++i) {
        int c = lane + 32 * i;
        y[c] = (v[i] - mean) * inv * g[c] + be[c];
    }
}

// ---------------- WMMA GEMM: Out[M,512] = A[M,512] (f32->f16) @ W[512,512] ----------------
// W is pre-transposed f16: Wt[n*512 + k].
// 256 threads = 8 waves; block tile 128Mx64N; wave tile 32Mx32N = 4 WMMA accumulators.
// Double-buffered LDS, one barrier per K-step, 128-bit staging, prefetch 2 steps ahead.
// act: 0 = none, 1 = +bias, 2 = +bias & relu
__global__ __launch_bounds__(256) void gemm512_wmma(const float* __restrict__ A,
                                                    const _Float16* __restrict__ Wt,
                                                    float* __restrict__ Out,
                                                    const float* __restrict__ bias,
                                                    int act) {
    __shared__ alignas(16) _Float16 lsA[2][128 * 32];   // [buf][row][k]
    __shared__ alignas(16) _Float16 lsB[2][64 * 32];    // [buf][n][k]
    const int tid  = threadIdx.x;
    const int lane = tid & 31;
    const int w    = tid >> 5;
    const int wr   = w >> 1, wc = w & 1;    // wave tile origin: (m0+wr*32, n0+wc*32)
    const int m0   = blockIdx.x * 128;
    const int n0   = blockIdx.y * 64;
    const int g    = lane >> 4;             // half-wave group
    const int lr   = lane & 15;

    // staging assignment (all contiguous 128-bit accesses)
    const int rA  = tid >> 1;               // 0..127
    const int kcA = (tid & 1) * 16;         // 0 or 16 (16 f32 per thread)
    const int nB  = tid >> 2;               // 0..63
    const int kcB = (tid & 3) * 8;          // 0,8,16,24 (8 f16 per thread)
    const float*    Abase = A  + (size_t)(m0 + rA) * CC + kcA;
    const _Float16* Bbase = Wt + (size_t)(n0 + nB) * CC + kcB;

    v8f c00 = {}, c01 = {}, c10 = {}, c11 = {};

    // prologue: stage K-tile 0 into buffer 0
    {
        f4 a0 = *(const f4*)(Abase);
        f4 a1 = *(const f4*)(Abase + 4);
        f4 a2 = *(const f4*)(Abase + 8);
        f4 a3 = *(const f4*)(Abase + 12);
        h8 x0, x1;
#pragma unroll
        for (int j = 0; j < 4; ++j) {
            x0[j] = (_Float16)a0[j]; x0[j + 4] = (_Float16)a1[j];
            x1[j] = (_Float16)a2[j]; x1[j + 4] = (_Float16)a3[j];
        }
        *(h8*)&lsA[0][rA * 32 + kcA]     = x0;
        *(h8*)&lsA[0][rA * 32 + kcA + 8] = x1;
        *(h8*)&lsB[0][nB * 32 + kcB]     = *(const h8*)(Bbase);
    }

    for (int step = 0; step < 16; ++step) {
        const int pb = step & 1;
        __syncthreads();

        const bool more = (step + 1) < 16;
        const int  kk   = (step + 1) * 32;

        // prefetch the tile after next into L2/WGP caches
        if (step + 2 < 16) {
            __builtin_prefetch(Abase + (step + 2) * 32, 0, 1);
            __builtin_prefetch(Bbase + (step + 2) * 32, 0, 1);
        }

        // issue global loads of next K-tile into registers (overlap WMMAs below)
        f4 pa0, pa1, pa2, pa3; h8 pbv;
        if (more) {
            pa0 = *(const f4*)(Abase + kk);
            pa1 = *(const f4*)(Abase + kk + 4);
            pa2 = *(const f4*)(Abase + kk + 8);
            pa3 = *(const f4*)(Abase + kk + 12);
            pbv = *(const h8*)(Bbase + kk);
        }

        // A fragments 16x32 (ISA 7.12.2: lanes 0-15 -> K 0..7,16..23 ; lanes 16-31 -> K 8..15,24..31)
        v16h a0, a1;
#pragma unroll
        for (int j = 0; j < 8; ++j) {
            int k = (j < 4) ? (8 * g + 2 * j) : (16 + 8 * g + 2 * (j - 4));
            h2 p0 = *reinterpret_cast<const h2*>(&lsA[pb][(wr * 32 + lr) * 32 + k]);
            a0[2 * j] = p0[0]; a0[2 * j + 1] = p0[1];
            h2 p1 = *reinterpret_cast<const h2*>(&lsA[pb][(wr * 32 + 16 + lr) * 32 + k]);
            a1[2 * j] = p1[0]; a1[2 * j + 1] = p1[1];
        }
        // B fragments 32x16 (lanes 0-15 -> K 0..7,16..23 ; lanes 16-31 -> K 8..15,24..31)
        v16h b0, b1;
#pragma unroll
        for (int j = 0; j < 8; ++j) {
            int k = (j < 4) ? (8 * g + 2 * j) : (16 + 8 * g + 2 * (j - 4));
            h2 q0 = *reinterpret_cast<const h2*>(&lsB[pb][(wc * 32 + lr) * 32 + k]);
            b0[2 * j] = q0[0]; b0[2 * j + 1] = q0[1];
            h2 q1 = *reinterpret_cast<const h2*>(&lsB[pb][(wc * 32 + 16 + lr) * 32 + k]);
            b1[2 * j] = q1[0]; b1[2 * j + 1] = q1[1];
        }

        c00 = __builtin_amdgcn_wmma_f32_16x16x32_f16(false, a0, false, b0, (short)0, c00, false, false);
        c01 = __builtin_amdgcn_wmma_f32_16x16x32_f16(false, a0, false, b1, (short)0, c01, false, false);
        c10 = __builtin_amdgcn_wmma_f32_16x16x32_f16(false, a1, false, b0, (short)0, c10, false, false);
        c11 = __builtin_amdgcn_wmma_f32_16x16x32_f16(false, a1, false, b1, (short)0, c11, false, false);

        // commit prefetched tile into the other buffer
        if (more) {
            const int qb = pb ^ 1;
            h8 x0, x1;
#pragma unroll
            for (int j = 0; j < 4; ++j) {
                x0[j] = (_Float16)pa0[j]; x0[j + 4] = (_Float16)pa1[j];
                x1[j] = (_Float16)pa2[j]; x1[j + 4] = (_Float16)pa3[j];
            }
            *(h8*)&lsA[qb][rA * 32 + kcA]     = x0;
            *(h8*)&lsA[qb][rA * 32 + kcA + 8] = x1;
            *(h8*)&lsB[qb][nB * 32 + kcB]     = pbv;
        }
    }

    // epilogue: C/D layout -> VGPR r: M = r + 8*g, N = lane&15
#pragma unroll
    for (int r = 0; r < 8; ++r) {
        int row0 = m0 + wr * 32 + r + 8 * g;
        int row1 = row0 + 16;
        int col0 = n0 + wc * 32 + lr;
        int col1 = col0 + 16;
        float v00 = c00[r], v01 = c01[r], v10 = c10[r], v11 = c11[r];
        if (act) {
            float bc0 = bias[col0], bc1 = bias[col1];
            v00 += bc0; v01 += bc1; v10 += bc0; v11 += bc1;
            if (act == 2) {
                v00 = fmaxf(v00, 0.f); v01 = fmaxf(v01, 0.f);
                v10 = fmaxf(v10, 0.f); v11 = fmaxf(v11, 0.f);
            }
        }
        Out[(size_t)row0 * CC + col0] = v00;
        Out[(size_t)row0 * CC + col1] = v01;
        Out[(size_t)row1 * CC + col0] = v10;
        Out[(size_t)row1 * CC + col1] = v11;
    }
}

// ---------------- fused causal attention (flash-style, no score tensor) ----------------
__global__ __launch_bounds__(256) void attn_kernel(const float* __restrict__ Q,
                                                   const float* __restrict__ K,
                                                   const float* __restrict__ V,
                                                   float* __restrict__ O) {
    __shared__ float Kt[32][HD];
    __shared__ float Vt[32][HD];
    int b = blockIdx.x >> 3;
    int h = blockIdx.x & 7;
    int t = threadIdx.x;
    bool active = (t < TT);
    size_t base = ((size_t)(b * TT + (active ? t : 0)) * CC) + h * HD;

    float qv[HD], acc[HD];
    float m = -1e30f, l = 0.f;
#pragma unroll
    for (int d = 0; d < HD; ++d) { qv[d] = active ? Q[base + d] : 0.f; acc[d] = 0.f; }

    for (int st = 0; st < TT; st += 32) {
#pragma unroll
        for (int i = 0; i < 8; ++i) {
            int idx = threadIdx.x + i * 256;
            int sr = idx >> 6, d = idx & 63;
            int s = st + sr;
            float kv = 0.f, vv = 0.f;
            if (s < TT) {
                size_t a2 = ((size_t)(b * TT + s) * CC) + h * HD + d;
                kv = K[a2]; vv = V[a2];
            }
            Kt[sr][d] = kv; Vt[sr][d] = vv;
        }
        __syncthreads();
        if (active) {
            int smax = t - st + 1;              // causal: s <= t
            if (smax > 32) smax = 32;
            for (int sr = 0; sr < smax; ++sr) {
                float dot = 0.f;
#pragma unroll
                for (int d = 0; d < HD; ++d) dot += qv[d] * Kt[sr][d];
                float sc   = dot * 0.125f;      // 1/sqrt(64)
                float nm   = fmaxf(m, sc);
                float sfac = __expf(m - nm);
                float p    = __expf(sc - nm);
                l = l * sfac + p;
#pragma unroll
                for (int d = 0; d < HD; ++d) acc[d] = acc[d] * sfac + p * Vt[sr][d];
                m = nm;
            }
        }
        __syncthreads();
    }
    if (active) {
        float inv = 1.f / l;
#pragma unroll
        for (int d = 0; d < HD; ++d) O[base + d] = acc[d] * inv;
    }
}

// ---------------- per-sample k-means, all state in LDS ----------------
__global__ __launch_bounds__(256) void kmeans_kernel(const float* __restrict__ X,
                                                     float* __restrict__ outIds) {
    __shared__ float cent[NCLUS][CC];   // 20 KB
    __shared__ float sums[NCLUS][CC];   // 20 KB
    __shared__ float cnt[NCLUS];
    __shared__ int   asg[TT];
    int b = blockIdx.x;
    const float* x = X + (size_t)b * TT * CC;
    int tid = threadIdx.x;

    for (int idx = tid; idx < NCLUS * CC; idx += 256) {
        int k = idx >> 9, c = idx & (CC - 1);
        cent[k][c] = x[(size_t)(k * (TT / NCLUS)) * CC + c];
    }
    __syncthreads();

    for (int iter = 0; iter <= KM_IT; ++iter) {
        if (tid < TT) {
            float d[NCLUS];
#pragma unroll
            for (int k = 0; k < NCLUS; ++k) d[k] = 0.f;
            for (int c0 = 0; c0 < CC; c0 += 16) {
                float xv[16];
#pragma unroll
                for (int j = 0; j < 16; ++j) xv[j] = x[(size_t)tid * CC + c0 + j];
#pragma unroll
                for (int k = 0; k < NCLUS; ++k) {
                    float s = 0.f;
#pragma unroll
                    for (int j = 0; j < 16; ++j) {
                        float df = xv[j] - cent[k][c0 + j];
                        s += df * df;
                    }
                    d[k] += s;
                }
            }
            int best = 0; float bd = d[0];
#pragma unroll
            for (int k = 1; k < NCLUS; ++k) if (d[k] < bd) { bd = d[k]; best = k; }
            asg[tid] = best;
        }
        __syncthreads();
        if (iter == KM_IT) break;

        for (int idx = tid; idx < NCLUS * CC; idx += 256) sums[idx >> 9][idx & (CC - 1)] = 0.f;
        if (tid < NCLUS) cnt[tid] = 0.f;
        __syncthreads();
        if (tid < TT) {
            int k = asg[tid];
            atomicAdd(&cnt[k], 1.f);
            for (int c = 0; c < CC; ++c) atomicAdd(&sums[k][c], x[(size_t)tid * CC + c]);
        }
        __syncthreads();
        for (int idx = tid; idx < NCLUS * CC; idx += 256) {
            int k = idx >> 9, c = idx & (CC - 1);
            if (cnt[k] > 0.f) cent[k][c] = sums[k][c] / cnt[k];
        }
        __syncthreads();
    }
    if (tid < TT) outIds[(size_t)b * TT + tid] = (float)asg[tid];
}

// ---------------- final pos/neg logits: one wave32 per (b,t) ----------------
__global__ __launch_bounds__(256) void logits_kernel(const float* __restrict__ out,
                                                     const int* __restrict__ posS,
                                                     const int* __restrict__ negS,
                                                     const int* __restrict__ uids,
                                                     const float* __restrict__ item_emb,
                                                     const float* __restrict__ user_emb,
                                                     float* __restrict__ dout) {
    int w = threadIdx.x >> 5, lane = threadIdx.x & 31;
    int bt = blockIdx.x * 8 + w;
    int b  = bt / TT;
    const float* o = out + (size_t)bt * CC;
    int pi = posS[bt], ni = negS[bt], ui = uids[b];
    float ps = 0.f, ns = 0.f;
#pragma unroll
    for (int i = 0; i < 16; ++i) {
        int c = lane + 32 * i;
        float ov = o[c];
        float pe, ne;
        if (c < 256) {
            pe = item_emb[(size_t)pi * 256 + c];
            ne = item_emb[(size_t)ni * 256 + c];
        } else {
            float ue = user_emb[(size_t)ui * 256 + (c - 256)];
            pe = ue; ne = ue;
        }
        ps += ov * pe; ns += ov * ne;
    }
#pragma unroll
    for (int off = 16; off > 0; off >>= 1) {
        ps += __shfl_xor(ps, off, 32);
        ns += __shfl_xor(ns, off, 32);
    }
    if (lane == 0) { dout[bt] = ps; dout[BT + bt] = ns; }
}

// ---------------- host-side orchestration ----------------
extern "C" void kernel_launch(void* const* d_in, const int* in_sizes, int n_in,
                              void* d_out, int out_size, void* d_ws, size_t ws_size,
                              hipStream_t stream) {
    (void)in_sizes; (void)n_in; (void)out_size; (void)ws_size;
    const int*   uids     = (const int*)d_in[0];
    const int*   seq      = (const int*)d_in[1];
    const int*   posS     = (const int*)d_in[2];
    const int*   negS     = (const int*)d_in[3];
    const float* item_emb = (const float*)d_in[4];
    const float* user_emb = (const float*)d_in[5];
    const float* pos_emb  = (const float*)d_in[6];
    const float* ln_g     = (const float*)d_in[7];
    const float* ln_b     = (const float*)d_in[8];

    float* ws = (float*)d_ws;
    float* xf = ws;                 // current activation x
    float* tA = ws + 1 * BTC;       // ln1 output (q residual)
    float* Qb = ws + 2 * BTC;
    float* Kb = ws + 3 * BTC;
    float* Vb = ws + 4 * BTC;
    float* Ob = ws + 5 * BTC;       // attention context
    float* tB = ws + 6 * BTC;       // Wo output
    _Float16* Wh = (_Float16*)(ws + 7 * BTC);   // 12 x 512x512 f16 transposed weights
    const size_t MS = (size_t)CC * CC;

    // 1. convert + transpose all projection/FFN weights to f16 [n][k]
    dim3 tgrid(CC / 32, CC / 32);
    for (int blk = 0; blk < 2; ++blk) {
        int base = 9 + blk * 12;
        const int wsel[6] = {0, 1, 2, 3, 4, 6};   // Wq,Wk,Wv,Wo,W1,W2
        for (int j = 0; j < 6; ++j)
            cvt_f16t_kernel<<<tgrid, 256, 0, stream>>>((const float*)d_in[base + wsel[j]],
                                                       Wh + (size_t)(blk * 6 + j) * MS);
    }

    // 2. embeddings (pre-mask x is what k-means sees)
    embed_kernel<<<(int)(BTC / 256), 256, 0, stream>>>(seq, uids, item_emb, user_emb, pos_emb, xf);

    // 3. k-means -> cluster ids (3rd output)
    kmeans_kernel<<<BB, 256, 0, stream>>>(xf, (float*)d_out + 2 * BT);

    // 4. timeline mask
    mask_kernel<<<(int)(BTC / 256), 256, 0, stream>>>(xf, seq);

    // 5. transformer blocks
    dim3 ggrid(BT / 128, CC / 64);
    for (int blk = 0; blk < 2; ++blk) {
        int base = 9 + blk * 12;
        const float* b1v  = (const float*)d_in[base + 5];
        const float* b2v  = (const float*)d_in[base + 7];
        const float* ln1g = (const float*)d_in[base + 8];
        const float* ln1b = (const float*)d_in[base + 9];
        const float* ln2g = (const float*)d_in[base + 10];
        const float* ln2b = (const float*)d_in[base + 11];
        const _Float16* Wq = Wh + (size_t)(blk * 6 + 0) * MS;
        const _Float16* Wk = Wh + (size_t)(blk * 6 + 1) * MS;
        const _Float16* Wv = Wh + (size_t)(blk * 6 + 2) * MS;
        const _Float16* Wo = Wh + (size_t)(blk * 6 + 3) * MS;
        const _Float16* W1 = Wh + (size_t)(blk * 6 + 4) * MS;
        const _Float16* W2 = Wh + (size_t)(blk * 6 + 5) * MS;

        ln_kernel<<<BT / 8, 256, 0, stream>>>(xf, tA, ln1g, ln1b);
        gemm512_wmma<<<ggrid, 256, 0, stream>>>(tA, Wq, Qb, nullptr, 0);
        gemm512_wmma<<<ggrid, 256, 0, stream>>>(xf, Wk, Kb, nullptr, 0);
        gemm512_wmma<<<ggrid, 256, 0, stream>>>(xf, Wv, Vb, nullptr, 0);
        attn_kernel<<<BB * NH, 256, 0, stream>>>(Qb, Kb, Vb, Ob);
        gemm512_wmma<<<ggrid, 256, 0, stream>>>(Ob, Wo, tB, nullptr, 0);
        addmask_kernel<<<(int)(BTC / 256), 256, 0, stream>>>(tA, tB, xf, seq);  // x=(q+o)*mask
        ln_kernel<<<BT / 8, 256, 0, stream>>>(xf, Qb, ln2g, ln2b);               // reuse Qb
        gemm512_wmma<<<ggrid, 256, 0, stream>>>(Qb, W1, Kb, b1v, 2);             // relu(.W1+b1)
        gemm512_wmma<<<ggrid, 256, 0, stream>>>(Kb, W2, Vb, b2v, 1);             // .W2+b2
        addmask_kernel<<<(int)(BTC / 256), 256, 0, stream>>>(xf, Vb, xf, seq);   // x=(x+y)*mask
    }

    // 6. final layernorm + logits
    ln_kernel<<<BT / 8, 256, 0, stream>>>(xf, tA, ln_g, ln_b);
    logits_kernel<<<BT / 8, 256, 0, stream>>>(tA, posS, negS, uids, item_emb, user_emb,
                                              (float*)d_out);
}
